// HierarchicalClassifier_9758165697076
// MI455X (gfx1250) — matrix operand order repaired and checked by reading, not compiled, not run on hardware
//
#include <hip/hip_runtime.h>
#include <hip/hip_bf16.h>
#include <math.h>

typedef __attribute__((ext_vector_type(2)))  float  v2f;
typedef __attribute__((ext_vector_type(8)))  float  v8f;
typedef __attribute__((ext_vector_type(16))) __bf16 v16bf;

#define B_   16
#define C_   64
#define HW_  4096
#define CQ_  8
#define NC_  6

__device__ __forceinline__ unsigned short f2bf(float x) {
  unsigned int u = __builtin_bit_cast(unsigned int, x);
  unsigned int r = (u + 0x7FFFu + ((u >> 16) & 1u)) >> 16;  // RNE
  return (unsigned short)r;
}

// ---------------------------------------------------------------------------
// Kernel 1: Q/K/V 1x1 convs. Q plain (pre-scaled by 1/sqrt(8)); K stored as
// f32 16x16x4 WMMA A-fragments; V stored as bf16 16x16x32 WMMA A-fragments.
// ---------------------------------------------------------------------------
__global__ __launch_bounds__(256) void qkv_kernel(
    const float* __restrict__ f,
    const float* __restrict__ qw, const float* __restrict__ qb,
    const float* __restrict__ kw, const float* __restrict__ kb,
    const float* __restrict__ vw, const float* __restrict__ vb,
    float* __restrict__ Qp, float* __restrict__ Kf,
    unsigned short* __restrict__ Vf)
{
  const int TOT = B_ * (CQ_ + CQ_ + C_) * HW_;
  int t = blockIdx.x * blockDim.x + threadIdx.x;
  if (t >= TOT) return;
  int hw  = t & (HW_ - 1);
  int tmp = t >> 12;
  int ri  = tmp % 80;
  int b   = tmp / 80;

  const float* fb = f + (size_t)b * C_ * HW_ + hw;
  const float* wrow;
  float bias;
  if (ri < 8)       { wrow = qw + ri * C_;        bias = qb[ri];      }
  else if (ri < 16) { wrow = kw + (ri - 8) * C_;  bias = kb[ri - 8];  }
  else              { wrow = vw + (ri - 16) * C_; bias = vb[ri - 16]; }

  float acc = bias;
#pragma unroll 8
  for (int ci = 0; ci < C_; ++ci) acc += wrow[ci] * fb[(size_t)ci * HW_];

  if (ri < 8) {
    Qp[((size_t)b * 8 + ri) * HW_ + hw] = acc * 0.35355339059327373f; // /sqrt(8)
  } else if (ri < 16) {
    // K fragment: per 16-key tile kt, frag f (d=4f..4f+3), lane=(h<<4)|key%16,
    // VGPR j holds K[d=4f+2h+j][key]
    int d = ri - 8;
    int kt = hw >> 4, m = hw & 15;
    int fi = d >> 2, rem = d & 3, h = rem >> 1, j = rem & 1;
    int lane = (h << 4) | m;
    size_t idx = ((((size_t)b * 256 + kt) * 2 + fi) * 32 + lane) * 2 + j;
    Kf[idx] = acc;
  } else {
    // V fragment (bf16 A 16x32): per 32-key tile kt, cg=c/16, lane=(h<<4)|c%16,
    // elem j: key = 16h+j
    int c = ri - 16;
    int kt = hw >> 5, kk = hw & 31;
    int h, j;
    h = kk >> 4; j = kk & 15;
    int cg = c >> 4, m = c & 15;
    int lane = (h << 4) | m;
    size_t idx = ((((size_t)b * 128 + kt) * 4 + cg) * 32 + lane) * 16 + j;
    Vf[idx] = f2bf(acc);
  }
}

// ---------------------------------------------------------------------------
// Kernel 2: flash attention. One wave = 16 queries x all 64 channels.
// 64 keys per iteration: S^T via 8x v_wmma_f32_16x16x4_f32, one online-softmax
// update, O^T += V x P^T via 8x v_wmma_f32_16x16x32_bf16.
// ---------------------------------------------------------------------------
__global__ __launch_bounds__(256) void attn_kernel(
    const float* __restrict__ Qp,
    const float* __restrict__ Kff,
    const unsigned short* __restrict__ Vff,
    float* __restrict__ att)
{
  const int lane = threadIdx.x & 31;
  const int wid  = (blockIdx.x * blockDim.x + threadIdx.x) >> 5;
  const int b  = wid >> 8;     // 256 query-tiles per batch
  const int qt = wid & 255;
  const int qb = qt << 4;
  const int h  = lane >> 4;
  const int n  = lane & 15;

  // Q as B-fragment of 16x16x4 f32 (K=d rows, N=query cols): VGPR j = Q[2h+j][q]
  const float* Q = Qp + (size_t)b * 8 * HW_;
  v2f qf0, qf1;
  qf0.x = Q[(2 * h + 0) * HW_ + qb + n];
  qf0.y = Q[(2 * h + 1) * HW_ + qb + n];
  qf1.x = Q[(4 + 2 * h + 0) * HW_ + qb + n];
  qf1.y = Q[(4 + 2 * h + 1) * HW_ + qb + n];

  const v2f*   kf = (const v2f*)Kff;
  const v16bf* vf = (const v16bf*)Vff;

  v8f o0 = {0,0,0,0,0,0,0,0}, o1 = {0,0,0,0,0,0,0,0};
  v8f o2 = {0,0,0,0,0,0,0,0}, o3 = {0,0,0,0,0,0,0,0};
  const v8f zero = {0,0,0,0,0,0,0,0};

  float mrow = -3.0e38f;
  float lsum = 0.0f;

  for (int kt = 0; kt < 64; ++kt) {   // 64 keys / iteration
    // ----- S^T tiles (M=key, N=query), four 16-key subtiles -----
    size_t kb4 = ((size_t)b * 256 + 4 * kt) * 2;
    v2f a0 = kf[(kb4 + 0) * 32 + lane];
    v2f a1 = kf[(kb4 + 1) * 32 + lane];
    v2f a2 = kf[(kb4 + 2) * 32 + lane];
    v2f a3 = kf[(kb4 + 3) * 32 + lane];
    v2f a4 = kf[(kb4 + 4) * 32 + lane];
    v2f a5 = kf[(kb4 + 5) * 32 + lane];
    v2f a6 = kf[(kb4 + 6) * 32 + lane];
    v2f a7 = kf[(kb4 + 7) * 32 + lane];
    v8f s0 = __builtin_amdgcn_wmma_f32_16x16x4_f32(false, a0, false, qf0,
                                                   (short)0, zero, false, false);
    s0 = __builtin_amdgcn_wmma_f32_16x16x4_f32(false, a1, false, qf1,
                                               (short)0, s0, false, false);
    v8f s1 = __builtin_amdgcn_wmma_f32_16x16x4_f32(false, a2, false, qf0,
                                                   (short)0, zero, false, false);
    s1 = __builtin_amdgcn_wmma_f32_16x16x4_f32(false, a3, false, qf1,
                                               (short)0, s1, false, false);
    v8f s2 = __builtin_amdgcn_wmma_f32_16x16x4_f32(false, a4, false, qf0,
                                                   (short)0, zero, false, false);
    s2 = __builtin_amdgcn_wmma_f32_16x16x4_f32(false, a5, false, qf1,
                                               (short)0, s2, false, false);
    v8f s3 = __builtin_amdgcn_wmma_f32_16x16x4_f32(false, a6, false, qf0,
                                                   (short)0, zero, false, false);
    s3 = __builtin_amdgcn_wmma_f32_16x16x4_f32(false, a7, false, qf1,
                                               (short)0, s3, false, false);

    // ----- one online-softmax update for all 64 keys -----
    float mt = mrow;
#pragma unroll
    for (int r = 0; r < 8; ++r) {
      mt = fmaxf(fmaxf(mt, s0[r]), s1[r]);
      mt = fmaxf(fmaxf(mt, s2[r]), s3[r]);
    }
    mt = fmaxf(mt, __shfl_xor(mt, 16, 32));
    float cf = __expf(mrow - mt);
    mrow = mt;
    lsum *= cf;
#pragma unroll
    for (int r = 0; r < 8; ++r) {
      o0[r] *= cf; o1[r] *= cf; o2[r] *= cf; o3[r] *= cf;
    }
    float p0[8], p1[8], p2[8], p3[8];
#pragma unroll
    for (int r = 0; r < 8; ++r) {
      p0[r] = __expf(s0[r] - mrow);
      p1[r] = __expf(s1[r] - mrow);
      p2[r] = __expf(s2[r] - mrow);
      p3[r] = __expf(s3[r] - mrow);
      lsum += (p0[r] + p1[r]) + (p2[r] + p3[r]);
    }

    // ----- build P^T bf16 B-fragments (32k x 16q), two key-halves -----
    // lane (h,n) elem j -> key 16h+j within the 32-key half
    union { unsigned int u[8]; v16bf v; } bun0, bun1;
#pragma unroll
    for (int i = 0; i < 4; ++i) {
      // half 0: subtiles p0 (keys 0-15), p1 (keys 16-31)
      float l0a = h ? __shfl_xor(p1[2*i],     16, 32) : p0[2*i];
      float l0b = h ? __shfl_xor(p1[2*i + 1], 16, 32) : p0[2*i + 1];
      float h0a = h ? p1[2*i]     : __shfl_xor(p0[2*i],     16, 32);
      float h0b = h ? p1[2*i + 1] : __shfl_xor(p0[2*i + 1], 16, 32);
      bun0.u[i]     = (unsigned)f2bf(l0a) | ((unsigned)f2bf(l0b) << 16);
      bun0.u[4 + i] = (unsigned)f2bf(h0a) | ((unsigned)f2bf(h0b) << 16);
      // half 1: subtiles p2 (keys 32-47), p3 (keys 48-63)
      float l1a = h ? __shfl_xor(p3[2*i],     16, 32) : p2[2*i];
      float l1b = h ? __shfl_xor(p3[2*i + 1], 16, 32) : p2[2*i + 1];
      float h1a = h ? p3[2*i]     : __shfl_xor(p2[2*i],     16, 32);
      float h1b = h ? p3[2*i + 1] : __shfl_xor(p2[2*i + 1], 16, 32);
      bun1.u[i]     = (unsigned)f2bf(l1a) | ((unsigned)f2bf(l1b) << 16);
      bun1.u[4 + i] = (unsigned)f2bf(h1a) | ((unsigned)f2bf(h1b) << 16);
    }

    // ----- O^T += V x P^T, 4 channel groups x 2 key-halves -----
    size_t vb4 = ((size_t)b * 128 + 2 * kt) * 4;
    o0 = __builtin_amdgcn_wmma_f32_16x16x32_bf16(false, vf[(vb4 + 0) * 32 + lane],
                                                 false, bun0.v, (short)0, o0, false, false);
    o1 = __builtin_amdgcn_wmma_f32_16x16x32_bf16(false, vf[(vb4 + 1) * 32 + lane],
                                                 false, bun0.v, (short)0, o1, false, false);
    o2 = __builtin_amdgcn_wmma_f32_16x16x32_bf16(false, vf[(vb4 + 2) * 32 + lane],
                                                 false, bun0.v, (short)0, o2, false, false);
    o3 = __builtin_amdgcn_wmma_f32_16x16x32_bf16(false, vf[(vb4 + 3) * 32 + lane],
                                                 false, bun0.v, (short)0, o3, false, false);
    o0 = __builtin_amdgcn_wmma_f32_16x16x32_bf16(false, vf[(vb4 + 4) * 32 + lane],
                                                 false, bun1.v, (short)0, o0, false, false);
    o1 = __builtin_amdgcn_wmma_f32_16x16x32_bf16(false, vf[(vb4 + 5) * 32 + lane],
                                                 false, bun1.v, (short)0, o1, false, false);
    o2 = __builtin_amdgcn_wmma_f32_16x16x32_bf16(false, vf[(vb4 + 6) * 32 + lane],
                                                 false, bun1.v, (short)0, o2, false, false);
    o3 = __builtin_amdgcn_wmma_f32_16x16x32_bf16(false, vf[(vb4 + 7) * 32 + lane],
                                                 false, bun1.v, (short)0, o3, false, false);
  }

  lsum += __shfl_xor(lsum, 16, 32);
  float inv = 1.0f / lsum;

  float* ab = att + (size_t)b * C_ * HW_ + qb + n;
#pragma unroll
  for (int r = 0; r < 8; ++r) {
    ab[(size_t)( 0 + r + 8 * h) * HW_] = o0[r] * inv;
    ab[(size_t)(16 + r + 8 * h) * HW_] = o1[r] * inv;
    ab[(size_t)(32 + r + 8 * h) * HW_] = o2[r] * inv;
    ab[(size_t)(48 + r + 8 * h) * HW_] = o3[r] * inv;
  }
}

// ---------------------------------------------------------------------------
// Kernel 3: context = features + scale * (ow @ attended + ob)
// ---------------------------------------------------------------------------
__global__ __launch_bounds__(256) void ctx_kernel(
    const float* __restrict__ f, const float* __restrict__ att,
    const float* __restrict__ ow, const float* __restrict__ ob,
    const float* __restrict__ scale, float* __restrict__ ctx)
{
  int t = blockIdx.x * blockDim.x + threadIdx.x;  // = (b*64+o)*4096 + hw
  if (t >= B_ * C_ * HW_) return;
  int hw = t & (HW_ - 1);
  int tmp = t >> 12;
  int o = tmp & 63;
  int b = tmp >> 6;
  const float* ar = att + (size_t)b * C_ * HW_ + hw;
  const float* wr = ow + o * C_;
  float acc = ob[o];
#pragma unroll 8
  for (int ci = 0; ci < C_; ++ci) acc += wr[ci] * ar[(size_t)ci * HW_];
  ctx[t] = f[t] + scale[0] * acc;
}

// ---------------------------------------------------------------------------
// Feature extractor (tiny): pools + 3x3 convs + BN/ReLU + heads
// ---------------------------------------------------------------------------
__global__ void pool8_kernel(const float* __restrict__ x, float* __restrict__ y) {
  int t = blockIdx.x * blockDim.x + threadIdx.x;  // (16,64,8,8)
  if (t >= B_ * C_ * 64) return;
  int xy = t & 63; int oy = xy >> 3, ox = xy & 7;
  int tmp = t >> 6; int c = tmp & 63; int b = tmp >> 6;
  const float* p = x + ((size_t)(b * 64 + c) * 64 + oy * 8) * 64 + ox * 8;
  float s = 0.f;
  for (int i = 0; i < 8; ++i)
    for (int j = 0; j < 8; ++j) s += p[i * 64 + j];
  y[t] = s * (1.0f / 64.0f);
}

__global__ void conv1_kernel(const float* __restrict__ x, const float* __restrict__ w,
                             const float* __restrict__ cb,
                             const float* __restrict__ g, const float* __restrict__ bb,
                             const float* __restrict__ bm, const float* __restrict__ bv,
                             float* __restrict__ y) {
  int t = blockIdx.x * blockDim.x + threadIdx.x;  // (16,128,8,8)
  if (t >= B_ * 128 * 64) return;
  int xy = t & 63; int oy = xy >> 3, ox = xy & 7;
  int tmp = t >> 6; int o = tmp & 127; int b = tmp >> 7;
  const float* wp = w + (size_t)o * 64 * 9;
  const float* xp = x + (size_t)b * 64 * 64;
  float acc = cb[o];
  for (int ci = 0; ci < 64; ++ci)
    for (int ky = -1; ky <= 1; ++ky) {
      int iy = oy + ky; if (iy < 0 || iy > 7) continue;
      for (int kx = -1; kx <= 1; ++kx) {
        int ix = ox + kx; if (ix < 0 || ix > 7) continue;
        acc += wp[ci * 9 + (ky + 1) * 3 + (kx + 1)] * xp[ci * 64 + iy * 8 + ix];
      }
    }
  float inv = g[o] * rsqrtf(bv[o] + 1e-5f);
  y[t] = fmaxf(acc * inv + (bb[o] - bm[o] * inv), 0.f);
}

__global__ void pool2_kernel(const float* __restrict__ x, float* __restrict__ y) {
  int t = blockIdx.x * blockDim.x + threadIdx.x;  // (16,128,4,4)
  if (t >= B_ * 128 * 16) return;
  int xy = t & 15; int oy = xy >> 2, ox = xy & 3;
  int tmp = t >> 4; int c = tmp & 127; int b = tmp >> 7;
  const float* p = x + ((size_t)(b * 128 + c) * 8 + oy * 2) * 8 + ox * 2;
  y[t] = (p[0] + p[1] + p[8] + p[9]) * 0.25f;
}

__global__ void conv2_kernel(const float* __restrict__ x, const float* __restrict__ w,
                             const float* __restrict__ cb,
                             const float* __restrict__ g, const float* __restrict__ bb,
                             const float* __restrict__ bm, const float* __restrict__ bv,
                             float* __restrict__ y) {
  int t = blockIdx.x * blockDim.x + threadIdx.x;  // (16,256,4,4)
  if (t >= B_ * 256 * 16) return;
  int xy = t & 15; int oy = xy >> 2, ox = xy & 3;
  int tmp = t >> 4; int o = tmp & 255; int b = tmp >> 8;
  const float* wp = w + (size_t)o * 128 * 9;
  const float* xp = x + (size_t)b * 128 * 16;
  float acc = cb[o];
  for (int ci = 0; ci < 128; ++ci)
    for (int ky = -1; ky <= 1; ++ky) {
      int iy = oy + ky; if (iy < 0 || iy > 3) continue;
      for (int kx = -1; kx <= 1; ++kx) {
        int ix = ox + kx; if (ix < 0 || ix > 3) continue;
        acc += wp[ci * 9 + (ky + 1) * 3 + (kx + 1)] * xp[ci * 16 + iy * 4 + ix];
      }
    }
  float inv = g[o] * rsqrtf(bv[o] + 1e-5f);
  y[t] = fmaxf(acc * inv + (bb[o] - bm[o] * inv), 0.f);
}

__global__ void pool3_kernel(const float* __restrict__ x, float* __restrict__ gfeat) {
  int t = blockIdx.x * blockDim.x + threadIdx.x;  // (16,256)
  if (t >= B_ * 256) return;
  const float* p = x + (size_t)t * 16;
  float s = 0.f;
  for (int i = 0; i < 16; ++i) s += p[i];
  gfeat[t] = s * (1.0f / 16.0f);
}

__global__ void heads_kernel(const float* __restrict__ gfeat,
                             const float* __restrict__ w1, const float* __restrict__ b1,
                             const float* __restrict__ w2, const float* __restrict__ b2,
                             float* __restrict__ out) {
  int b = blockIdx.x * blockDim.x + threadIdx.x;
  if (b >= B_) return;
  const float* g = gfeat + b * 256;
  float l1[NC_], p1[NC_], l2[NC_], p2[NC_];
  for (int o = 0; o < NC_; ++o) {
    float a = b1[o];
    for (int i = 0; i < 256; ++i) a += g[i] * w1[o * 256 + i];
    l1[o] = a;
  }
  float mx = l1[0];
  for (int o = 1; o < NC_; ++o) mx = fmaxf(mx, l1[o]);
  float s = 0.f;
  for (int o = 0; o < NC_; ++o) { p1[o] = __expf(l1[o] - mx); s += p1[o]; }
  for (int o = 0; o < NC_; ++o) p1[o] /= s;
  for (int o = 0; o < NC_; ++o) {
    float a = b2[o];
    for (int i = 0; i < 256; ++i) a += g[i] * w2[o * 262 + i];
    for (int j = 0; j < NC_; ++j) a += p1[j] * w2[o * 262 + 256 + j];
    l2[o] = a;
  }
  mx = l2[0];
  for (int o = 1; o < NC_; ++o) mx = fmaxf(mx, l2[o]);
  s = 0.f;
  for (int o = 0; o < NC_; ++o) { p2[o] = __expf(l2[o] - mx); s += p2[o]; }
  for (int o = 0; o < NC_; ++o) p2[o] /= s;
  for (int o = 0; o < NC_; ++o) {
    out[  0 + b * NC_ + o] = l1[o];
    out[ 96 + b * NC_ + o] = l2[o];
    out[192 + b * NC_ + o] = p1[o];
    out[288 + b * NC_ + o] = p2[o];
  }
}

// ---------------------------------------------------------------------------
extern "C" void kernel_launch(void* const* d_in, const int* in_sizes, int n_in,
                              void* d_out, int out_size, void* d_ws, size_t ws_size,
                              hipStream_t stream) {
  const float* features = (const float*)d_in[0];
  const float* qw = (const float*)d_in[1];
  const float* qb = (const float*)d_in[2];
  const float* kw = (const float*)d_in[3];
  const float* kb = (const float*)d_in[4];
  const float* vw = (const float*)d_in[5];
  const float* vb = (const float*)d_in[6];
  const float* ow = (const float*)d_in[7];
  const float* ob = (const float*)d_in[8];
  const float* scale = (const float*)d_in[9];
  const float* c1w = (const float*)d_in[10];
  const float* c1b = (const float*)d_in[11];
  const float* bn1g = (const float*)d_in[12];
  const float* bn1b = (const float*)d_in[13];
  const float* bn1m = (const float*)d_in[14];
  const float* bn1v = (const float*)d_in[15];
  const float* c2w = (const float*)d_in[16];
  const float* c2b = (const float*)d_in[17];
  const float* bn2g = (const float*)d_in[18];
  const float* bn2b = (const float*)d_in[19];
  const float* bn2m = (const float*)d_in[20];
  const float* bn2v = (const float*)d_in[21];
  const float* w1 = (const float*)d_in[22];
  const float* b1 = (const float*)d_in[23];
  const float* w2 = (const float*)d_in[24];
  const float* b2 = (const float*)d_in[25];

  char* ws = (char*)d_ws;
  float*          Qp   = (float*)(ws + 0);                  //  2 MB
  float*          Kf   = (float*)(ws + (2u << 20));         //  2 MB
  unsigned short* Vf   = (unsigned short*)(ws + (4u << 20));//  8 MB
  float*          att  = (float*)(ws + (12u << 20));        // 16 MB
  float*          pl1  = (float*)(ws + (28u << 20));        // 256 KB
  float*          x1   = (float*)(ws + (28u << 20) + (1u << 19));   // 512 KB
  float*          pl2  = (float*)(ws + (28u << 20) + (3u << 19));   // 128 KB
  float*          x2   = (float*)(ws + (28u << 20) + (7u << 18));   // 256 KB

  float* out = (float*)d_out;
  float* gfeat_out = out + 384;
  float* ctx_out   = out + 4480;

  // 1) Q/K/V projections + WMMA fragment swizzles
  {
    int tot = B_ * 80 * HW_;
    qkv_kernel<<<(tot + 255) / 256, 256, 0, stream>>>(
        features, qw, qb, kw, kb, vw, vb, Qp, Kf, Vf);
  }
  // 2) flash attention (16 queries x 64 channels per wave, 64 keys/iter)
  attn_kernel<<<512, 256, 0, stream>>>(Qp, Kf, Vf, att);
  // 3) output proj + residual -> context (into d_out)
  {
    int tot = B_ * C_ * HW_;
    ctx_kernel<<<tot / 256, 256, 0, stream>>>(features, att, ow, ob, scale, ctx_out);
  }
  // 4) feature extractor
  pool8_kernel<<<(B_ * C_ * 64) / 256, 256, 0, stream>>>(ctx_out, pl1);
  conv1_kernel<<<(B_ * 128 * 64) / 256, 256, 0, stream>>>(
      pl1, c1w, c1b, bn1g, bn1b, bn1m, bn1v, x1);
  pool2_kernel<<<(B_ * 128 * 16) / 256, 256, 0, stream>>>(x1, pl2);
  conv2_kernel<<<(B_ * 256 * 16) / 256, 256, 0, stream>>>(
      pl2, c2w, c2b, bn2g, bn2b, bn2m, bn2v, x2);
  pool3_kernel<<<(B_ * 256) / 256, 256, 0, stream>>>(x2, gfeat_out);
  // 5) hierarchical heads
  heads_kernel<<<1, 32, 0, stream>>>(gfeat_out, w1, b1, w2, b2, out);
}
